// New_GAU_73254962200675
// MI455X (gfx1250) — compile-verified, matrix operand order checked by reading, not run on hardware
//
#include <hip/hip_runtime.h>
#include <hip/hip_bf16.h>
#include <math.h>

typedef __bf16 bf16_t;
typedef __attribute__((ext_vector_type(8)))  __bf16 v8bf;
typedef __attribute__((ext_vector_type(16))) __bf16 v16bf;
typedef __attribute__((ext_vector_type(8)))  float  v8f;

#define BATCH 16
#define SEQ   2048
#define CH    256
#define TOK   (BATCH * SEQ)   // 32768 rows total

// ---------------------------------------------------------------------------
// WMMA helpers (CDNA5 wave32, v_wmma_f32_16x16x32_bf16)
// A fragment: lane (l&15) = row M; lanes <16 hold K {k0..k0+7, k0+16..k0+23},
// lanes >=16 hold K {k0+8..k0+15, k0+24..k0+31}. B mirrors with lane = col N;
// every B operand is stored transposed so its columns are contiguous.
// ---------------------------------------------------------------------------
__device__ __forceinline__ v16bf frag16(const bf16_t* tile, int stride, int lane, int k0) {
  const int r  = lane & 15;
  const int kb = k0 + ((lane >> 4) << 3);
  const bf16_t* p = tile + r * stride + kb;
  v8bf lo = *(const v8bf*)(p);        // K = kb .. kb+7       (16B load)
  v8bf hi = *(const v8bf*)(p + 16);   // K = kb+16 .. kb+23   (16B load)
  return __builtin_shufflevector(lo, hi, 0,1,2,3,4,5,6,7,8,9,10,11,12,13,14,15);
}

__device__ __forceinline__ v8f wmma_bf16(v16bf a, v16bf b, v8f c) {
  return __builtin_amdgcn_wmma_f32_16x16x32_bf16(false, a, false, b, (short)0, c, false, false);
}

__device__ __forceinline__ v8f zero8() { v8f z = {0.f,0.f,0.f,0.f,0.f,0.f,0.f,0.f}; return z; }

// ---------------------------------------------------------------------------
// Tensor Data Mover (clang-23 6-arg builtin). D# per CDNA5 ISA ch.8:
// group0: count=1 | lds_addr | global_addr[56:0] | type=2
// group1: data_size=2B | tensor_dim0/1 | tile_dim0/1 | dim0_stride
// 2D tile load, groups 2/3 zero (disabled).
// ---------------------------------------------------------------------------
#if __has_builtin(__builtin_amdgcn_tensor_load_to_lds) && __has_builtin(__builtin_amdgcn_s_wait_tensorcnt)
#define HAVE_TDM 1
typedef unsigned int u32x4 __attribute__((ext_vector_type(4)));
typedef int          i32x4 __attribute__((ext_vector_type(4)));
typedef int          i32x8 __attribute__((ext_vector_type(8)));

__device__ __forceinline__ void tdm_load_2d(const void* gptr, const void* lptr,
                                            unsigned tensor_d0, unsigned tensor_d1,
                                            unsigned tile_d0, unsigned tile_d1,
                                            unsigned long long stride0_elems) {
  const unsigned long long ga = (unsigned long long)(size_t)gptr;
  const unsigned lds = (unsigned)(size_t)lptr;   // low 32 bits of LDS aperture addr = LDS offset
  u32x4 g0;
  g0[0] = 1u;                                              // count=1 (valid user D#)
  g0[1] = lds;                                             // lds_addr
  g0[2] = (unsigned)(ga & 0xffffffffu);                    // global_addr[31:0]
  g0[3] = (unsigned)((ga >> 32) & 0x01ffffffu) | (2u << 30); // global_addr[56:32] | type=2
  i32x8 g1;
  g1[0] = (int)(1u << 16);                                 // wg_mask=0, data_size=2 bytes
  g1[1] = (int)((tensor_d0 & 0xffffu) << 16);              // tensor_dim0[15:0]
  g1[2] = (int)((tensor_d0 >> 16) | ((tensor_d1 & 0xffffu) << 16));
  g1[3] = (int)((tensor_d1 >> 16) | (tile_d0 << 16));      // tensor_dim1[31:16] | tile_dim0
  g1[4] = (int)(tile_d1 & 0xffffu);                        // tile_dim1, tile_dim2=0
  g1[5] = (int)(unsigned)(stride0_elems & 0xffffffffu);    // dim0_stride[31:0]
  g1[6] = (int)(unsigned)((stride0_elems >> 32) & 0xffffu);// dim0_stride[47:32], dim1_stride=0
  g1[7] = 0;
  const i32x4 z4 = {0, 0, 0, 0};
  const i32x8 z8 = {0, 0, 0, 0, 0, 0, 0, 0};
  __builtin_amdgcn_tensor_load_to_lds(g0, g1, z4, z4, z8, 0);
}
#else
#define HAVE_TDM 0
#endif

// ---------------------------------------------------------------------------
// Kernel 1: convert weights to bf16, transposed so B-columns are contiguous.
// ---------------------------------------------------------------------------
__global__ void k_prep_weights(const float* __restrict__ w_hidden,
                               const float* __restrict__ w_kv,
                               const float* __restrict__ w_proj,
                               bf16_t* __restrict__ WcT,
                               bf16_t* __restrict__ WpT) {
  const int i = blockIdx.x * blockDim.x + threadIdx.x;
  if (i < 768 * CH) {
    const int c = i >> 8, r = i & 255;
    const float v = (c < 512) ? w_hidden[r * 512 + c] : w_kv[r * 256 + (c - 512)];
    WcT[i] = (bf16_t)v;
  } else {
    const int j = i - 768 * CH;
    const int c = j >> 8, r = j & 255;
    WpT[j] = (bf16_t)w_proj[r * 256 + c];
  }
}

// ---------------------------------------------------------------------------
// Kernel 2: LayerNorm. One wave32 per 256-wide row; shuffle reductions.
// ---------------------------------------------------------------------------
__global__ __launch_bounds__(256) void k_layernorm(const float* __restrict__ x,
                                                   const float* __restrict__ lnw,
                                                   const float* __restrict__ lnb,
                                                   bf16_t* __restrict__ out) {
  const int lane = threadIdx.x & 31;
  const int wave = threadIdx.x >> 5;
  const int row  = blockIdx.x * 8 + wave;
  const float* xr = x + (size_t)row * CH + lane * 8;
  float v[8];
  float s = 0.f;
  #pragma unroll
  for (int i = 0; i < 8; ++i) { v[i] = xr[i]; s += v[i]; }
  #pragma unroll
  for (int off = 16; off >= 1; off >>= 1) s += __shfl_xor(s, off, 32);
  const float mu = s * (1.0f / CH);
  float q = 0.f;
  #pragma unroll
  for (int i = 0; i < 8; ++i) { const float d = v[i] - mu; q += d * d; }
  #pragma unroll
  for (int off = 16; off >= 1; off >>= 1) q += __shfl_xor(q, off, 32);
  const float rstd = rsqrtf(q * (1.0f / CH) + 1e-5f);
  v8bf o;
  #pragma unroll
  for (int i = 0; i < 8; ++i) {
    const int c = lane * 8 + i;
    o[i] = (bf16_t)((v[i] - mu) * rstd * lnw[c] + lnb[c]);
  }
  *(v8bf*)(out + (size_t)row * CH + lane * 8) = o;
}

// ---------------------------------------------------------------------------
// Kernel 3: fused input GEMM. O = normed(32768x256) @ Wc(256x768).
// Epilogue: cols [0,256)->silu->v ; [256,512)->silu->gate ;
//           [512,768)->Z-> q=Z*g0+b0, k=Z*g1+b1.
// ---------------------------------------------------------------------------
__global__ __launch_bounds__(256) void k_gemm_in(const bf16_t* __restrict__ nrm,
                                                 const bf16_t* __restrict__ WcT,
                                                 const float* __restrict__ b_hidden,
                                                 const float* __restrict__ gamma,
                                                 const float* __restrict__ beta,
                                                 bf16_t* __restrict__ vO,
                                                 bf16_t* __restrict__ gO,
                                                 bf16_t* __restrict__ qO,
                                                 bf16_t* __restrict__ kO) {
  const int lane = threadIdx.x & 31;
  const int wave = threadIdx.x >> 5;
  const int mt = blockIdx.x;
  const int nt = blockIdx.y * 8 + wave;
  const bf16_t* at = nrm + (size_t)(mt * 16) * CH;
  const bf16_t* bt = WcT + (size_t)(nt * 16) * CH;
  v8f acc = zero8();
  #pragma unroll
  for (int kk = 0; kk < 8; ++kk)
    acc = wmma_bf16(frag16(at, CH, lane, kk * 32), frag16(bt, CH, lane, kk * 32), acc);

  const int col   = nt * 16 + (lane & 15);
  const int rbase = mt * 16 + ((lane >> 4) << 3);
  #pragma unroll
  for (int i = 0; i < 8; ++i) {
    const int row = rbase + i;
    const float d = acc[i];
    if (col < 512) {
      const float h = d + b_hidden[col];
      const float s = h / (1.0f + __expf(-h));        // silu
      if (col < 256) vO[(size_t)row * CH + col]         = (bf16_t)s;
      else           gO[(size_t)row * CH + (col - 256)] = (bf16_t)s;
    } else {
      const int zc = col - 512;
      qO[(size_t)row * CH + zc] = (bf16_t)(d * gamma[zc]      + beta[zc]);
      kO[(size_t)row * CH + zc] = (bf16_t)(d * gamma[CH + zc] + beta[CH + zc]);
    }
  }
}

// ---------------------------------------------------------------------------
// Kernel 3b: one-time LDS-tiled transpose  vT[b][c][j] = v[b][j][c]
// 64x64 tiles, 256 threads, all global accesses 16B.
// ---------------------------------------------------------------------------
__global__ __launch_bounds__(256) void k_transpose_v(const bf16_t* __restrict__ v,
                                                     bf16_t* __restrict__ vT) {
  __shared__ __align__(16) bf16_t t[64][80];   // pad keeps 16B alignment, spreads banks
  const int b  = blockIdx.z;
  const int j0 = blockIdx.x * 64;
  const int c0 = blockIdx.y * 64;
  const int tid = threadIdx.x;
  {
    const int jl = tid >> 2, cl = (tid & 3) << 4;
    const bf16_t* src = v + ((size_t)b * SEQ + j0 + jl) * CH + c0 + cl;
    *(int4*)&t[jl][cl]     = *(const int4*)src;
    *(int4*)&t[jl][cl + 8] = *(const int4*)(src + 8);
  }
  __syncthreads();
  {
    const int cl = tid >> 2, jl = (tid & 3) << 4;
    union { bf16_t h[16]; int4 q[2]; } u;
    #pragma unroll
    for (int i = 0; i < 16; ++i) u.h[i] = t[jl + i][cl];
    int4* dst = (int4*)(vT + ((size_t)b * CH + c0 + cl) * SEQ + j0 + jl);
    dst[0] = u.q[0];
    dst[1] = u.q[1];
  }
}

// ---------------------------------------------------------------------------
// Kernel 4: GAU attention. Workgroup = 64 rows x 256 cols of one batch.
// Streams 32-column chunks; k and vT staged into double-buffered LDS by the
// Tensor Data Mover (overlapping DMA of chunk n+1 with WMMA on chunk n),
// falling back to cooperative 16B copies if the TDM builtin is unavailable.
// ---------------------------------------------------------------------------
__global__ __launch_bounds__(256) void k_attn(const bf16_t* __restrict__ qB,
                                              const bf16_t* __restrict__ kB,
                                              const bf16_t* __restrict__ vT,
                                              const bf16_t* __restrict__ gB,
                                              bf16_t* __restrict__ U) {
  __shared__ __align__(16) bf16_t sK [2][32 * CH];   // k chunk row-major [j][c]  2x16KB
  __shared__ __align__(16) bf16_t sVT[2][CH * 32];   // vT chunk [c][j]           2x16KB
  __shared__ __align__(16) bf16_t sS [64 * 32];      // relu^2 score chunk          4KB

  const int tid  = threadIdx.x;
  const int lane = tid & 31;
  const int wave = tid >> 5;
  const int rt = wave >> 1;
  const int jt = wave & 1;
  const int b  = blockIdx.y;
  const int r0 = blockIdx.x * 64;
  const size_t base = (size_t)b * SEQ * CH;
  const bf16_t* kBase  = kB + base;                 // [2048][256]
  const bf16_t* vTBase = vT + base;                 // [256][2048]

  // q A-fragments for this wave's 16 rows, full K=256 (held in VGPRs)
  v16bf qf[8];
  const bf16_t* qt = qB + base + (size_t)(r0 + rt * 16) * CH;
  #pragma unroll
  for (int kk = 0; kk < 8; ++kk) qf[kk] = frag16(qt, CH, lane, kk * 32);

  v8f acc[8];
  #pragma unroll
  for (int t = 0; t < 8; ++t) acc[t] = zero8();

#if HAVE_TDM
  if (wave == 0) {
    tdm_load_2d(kBase,  sK[0],  CH, 32, CH, 32, CH);          // 32 rows of 256
    tdm_load_2d(vTBase, sVT[0], SEQ, CH, 32, CH, SEQ);        // 256 rows of 32
  }
#else
  {  // prologue cooperative stage of chunk 0
    const int4* src = (const int4*)kBase;
    int4* dst = (int4*)sK[0];
    #pragma unroll
    for (int t = 0; t < 4; ++t) dst[tid + t * 256] = src[tid + t * 256];
    const int4* vsrc = (const int4*)(vTBase + (size_t)tid * SEQ);
    int4* vdst = (int4*)(sVT[0] + tid * 32);
    vdst[0] = vsrc[0];
    vdst[1] = vsrc[1];
  }
#endif

  for (int j0 = 0, it = 0; j0 < SEQ; j0 += 32, ++it) {
    const int cur = it & 1;
    const bool more = (j0 + 32) < SEQ;

    // ---- issue staging of NEXT chunk into the free buffer ----
#if HAVE_TDM
    if (wave == 0) {
      if (more) {
        tdm_load_2d(kBase + (size_t)(j0 + 32) * CH, sK[cur ^ 1], CH, 32, CH, 32, CH);
        tdm_load_2d(vTBase + (j0 + 32),             sVT[cur ^ 1], SEQ, CH, 32, CH, SEQ);
        __builtin_amdgcn_s_wait_tensorcnt(2);   // current chunk's 2 DMAs done (in-order)
      } else {
        __builtin_amdgcn_s_wait_tensorcnt(0);
      }
    }
#else
    if (more) {
      const int4* src = (const int4*)(kBase + (size_t)(j0 + 32) * CH);
      int4* dst = (int4*)sK[cur ^ 1];
      #pragma unroll
      for (int t = 0; t < 4; ++t) dst[tid + t * 256] = src[tid + t * 256];
      const int4* vsrc = (const int4*)(vTBase + (size_t)tid * SEQ + (j0 + 32));
      int4* vdst = (int4*)(sVT[cur ^ 1] + tid * 32);
      vdst[0] = vsrc[0];
      vdst[1] = vsrc[1];
    }
#endif
    __syncthreads();   // current buffer visible to all waves

    // ---- GEMM 1: S(16x16) = q_rows(16x256) . k_chunk(32x256)^T slice ----
    v8f s = zero8();
    #pragma unroll
    for (int kk = 0; kk < 8; ++kk)
      s = wmma_bf16(qf[kk], frag16(&sK[cur][jt * 16 * CH], CH, lane, kk * 32), s);

    // relu(x/n)^2 -> bf16 scores into LDS (re-shapes D-layout -> A-layout)
    const int scol   = jt * 16 + (lane & 15);
    const int srbase = rt * 16 + ((lane >> 4) << 3);
    #pragma unroll
    for (int i = 0; i < 8; ++i) {
      float xv = s[i] * (1.0f / SEQ);
      xv = fmaxf(xv, 0.0f);
      sS[(srbase + i) * 32 + scol] = (bf16_t)(xv * xv);
    }
    __syncthreads();

    // ---- GEMM 2: V(16x16 per ct) += S(16x32) . v_chunk(32x16) ----
    const v16bf af = frag16(&sS[rt * 16 * 32], 32, lane, 0);
    #pragma unroll
    for (int ct = 0; ct < 8; ++ct) {
      const int cg = jt * 8 + ct;
      acc[ct] = wmma_bf16(af, frag16(&sVT[cur][cg * 16 * 32], 32, lane, 0), acc[ct]);
    }
    __syncthreads();   // all reads of buffer `cur` done before it is re-filled
  }

  // ---- epilogue: U = V * gate (bf16) ----
  const int rb = r0 + rt * 16 + ((lane >> 4) << 3);
  #pragma unroll
  for (int ct = 0; ct < 8; ++ct) {
    const int cg  = jt * 8 + ct;
    const int col = cg * 16 + (lane & 15);
    #pragma unroll
    for (int i = 0; i < 8; ++i) {
      const size_t idx = base + (size_t)(rb + i) * CH + col;
      U[idx] = (bf16_t)(acc[ct][i] * (float)gB[idx]);
    }
  }
}

// ---------------------------------------------------------------------------
// Kernel 5: out = U(32768x256) @ w_proj(256x256) + b_proj + x   (f32 out)
// ---------------------------------------------------------------------------
__global__ __launch_bounds__(256) void k_gemm_out(const bf16_t* __restrict__ U,
                                                  const bf16_t* __restrict__ WpT,
                                                  const float* __restrict__ b_proj,
                                                  const float* __restrict__ x,
                                                  float* __restrict__ out) {
  const int lane = threadIdx.x & 31;
  const int wave = threadIdx.x >> 5;
  const int mt = blockIdx.x;
  const int nt = blockIdx.y * 8 + wave;
  const bf16_t* at = U   + (size_t)(mt * 16) * CH;
  const bf16_t* bt = WpT + (size_t)(nt * 16) * CH;
  v8f acc = zero8();
  #pragma unroll
  for (int kk = 0; kk < 8; ++kk)
    acc = wmma_bf16(frag16(at, CH, lane, kk * 32), frag16(bt, CH, lane, kk * 32), acc);

  const int col   = nt * 16 + (lane & 15);
  const int rbase = mt * 16 + ((lane >> 4) << 3);
  const float bp = b_proj[col];
  #pragma unroll
  for (int i = 0; i < 8; ++i) {
    const size_t idx = (size_t)(rbase + i) * CH + col;
    out[idx] = acc[i] + bp + x[idx];
  }
}

// ---------------------------------------------------------------------------
extern "C" void kernel_launch(void* const* d_in, const int* in_sizes, int n_in,
                              void* d_out, int out_size, void* d_ws, size_t ws_size,
                              hipStream_t stream) {
  const float* x        = (const float*)d_in[0];
  const float* ln_w     = (const float*)d_in[3];
  const float* ln_b     = (const float*)d_in[4];
  const float* w_hidden = (const float*)d_in[5];
  const float* b_hidden = (const float*)d_in[6];
  const float* w_kv     = (const float*)d_in[7];
  const float* gamma    = (const float*)d_in[8];
  const float* beta     = (const float*)d_in[9];
  const float* w_proj   = (const float*)d_in[10];
  const float* b_proj   = (const float*)d_in[11];
  float* out = (float*)d_out;

  // workspace carve-up (bf16 elements); total ~113 MB
  bf16_t* ws  = (bf16_t*)d_ws;
  const size_t PLANE = (size_t)TOK * CH;   // 8388608 elems
  bf16_t* nrm = ws;
  bf16_t* qb  = nrm + PLANE;
  bf16_t* kb  = qb  + PLANE;
  bf16_t* vb  = kb  + PLANE;
  bf16_t* vTb = vb  + PLANE;
  bf16_t* gb  = vTb + PLANE;
  bf16_t* Ub  = gb  + PLANE;
  bf16_t* WcT = Ub  + PLANE;               // 768*256
  bf16_t* WpT = WcT + (size_t)768 * CH;    // 256*256

  k_prep_weights<<<dim3(1024), dim3(256), 0, stream>>>(w_hidden, w_kv, w_proj, WcT, WpT);
  k_layernorm  <<<dim3(TOK / 8), dim3(256), 0, stream>>>(x, ln_w, ln_b, nrm);
  k_gemm_in    <<<dim3(TOK / 16, 6), dim3(256), 0, stream>>>(nrm, WcT, b_hidden, gamma, beta,
                                                             vb, gb, qb, kb);
  k_transpose_v<<<dim3(SEQ / 64, CH / 64, BATCH), dim3(256), 0, stream>>>(vb, vTb);
  k_attn       <<<dim3(SEQ / 64, BATCH), dim3(256), 0, stream>>>(qb, kb, vTb, gb, Ub);
  k_gemm_out   <<<dim3(TOK / 16, 2), dim3(256), 0, stream>>>(Ub, WpT, b_proj, x, out);
}
